// ResidualMLP_88622355186238
// MI455X (gfx1250) — matrix operand order repaired and checked by reading, not compiled, not run on hardware
//
#include <hip/hip_runtime.h>
#include <hip/hip_bf16.h>

typedef __attribute__((ext_vector_type(16))) __bf16 v16bf;
typedef __attribute__((ext_vector_type(8)))  float  v8f;

// ---------------- helpers ----------------

__device__ __forceinline__ float bf16r(float x) {
    return __bfloat162float(__float2bfloat16(x));
}

// MXINT8 fake-quant given shared block amax (block of 32 along K).
// q * 2^(e-6), |q|<=128 -> exactly representable in bf16.
__device__ __forceinline__ float mx_q(float v, float amax) {
    if (!(amax > 0.0f)) return 0.0f;
    float e     = floorf(log2f(amax));
    float scale = exp2f(e - 6.0f);            // shared_exp - (BITS-2), BITS=8
    float q     = rintf(v / scale);           // round-half-even like jnp.round
    q = fminf(fmaxf(q, -128.0f), 127.0f);
    return q * scale;
}

// CDNA5 async global->LDS DMA (ASYNCcnt-tracked), 16 bytes per lane.
// LDS operand: addrspace(3) byte offset (= low 32 bits of generic pointer on AMDGCN).
__device__ __forceinline__ void async_copy16(unsigned lds_off, const void* gptr) {
    asm volatile("global_load_async_to_lds_b128 %0, %1, off"
                 :
                 : "v"(lds_off), "v"((unsigned long long)(uintptr_t)gptr)
                 : "memory");
}
__device__ __forceinline__ void wait_async0() {
    asm volatile("s_wait_asynccnt 0" ::: "memory");
}

// ---------------- LayerNorm + MX quant (one block per token, H=2048) ----------------

__global__ __launch_bounds__(256)
void ln_quant_kernel(const float* __restrict__ in,
                     const float* __restrict__ w,
                     const float* __restrict__ b,
                     __hip_bfloat16* __restrict__ xq,
                     int H) {
    __shared__ float s_sum[256];
    __shared__ float s_sq[256];
    const int   t   = blockIdx.x;
    const int   tid = threadIdx.x;
    const size_t base = (size_t)t * H + (size_t)tid * 8;

    float x[8];
    float s = 0.0f, ss = 0.0f;
#pragma unroll
    for (int i = 0; i < 8; ++i) {
        x[i] = bf16r(in[base + i]);           // x = bf16_round(inputs)
        s  += x[i];
        ss += x[i] * x[i];
    }
    s_sum[tid] = s; s_sq[tid] = ss;
    __syncthreads();
    for (int off = 128; off > 0; off >>= 1) {
        if (tid < off) { s_sum[tid] += s_sum[tid + off]; s_sq[tid] += s_sq[tid + off]; }
        __syncthreads();
    }
    const float mu  = s_sum[0] / (float)H;
    const float var = s_sq[0] / (float)H - mu * mu;
    const float inv = rsqrtf(var + 1e-5f);

    float g[8];
    float amax = 0.0f;
#pragma unroll
    for (int i = 0; i < 8; ++i) {
        int c = tid * 8 + i;
        g[i] = bf16r((x[i] - mu) * inv * w[c] + b[c]);   // bf16-rounded LN output
        amax = fmaxf(amax, fabsf(g[i]));
    }
    // One MX block (32 elems) = 4 consecutive lanes * 8 elems
    amax = fmaxf(amax, __shfl_xor(amax, 1, 4));
    amax = fmaxf(amax, __shfl_xor(amax, 2, 4));
#pragma unroll
    for (int i = 0; i < 8; ++i)
        xq[base + i] = __float2bfloat16(mx_q(g[i], amax));
}

// ---------------- MX quant of fp32 weights (blocks of 32 along contiguous K) ----------------

__global__ __launch_bounds__(256)
void quant_rows_kernel(const float* __restrict__ W,
                       __hip_bfloat16* __restrict__ Wq,
                       size_t n) {
    size_t i0 = ((size_t)blockIdx.x * blockDim.x + threadIdx.x) * 8;
    if (i0 >= n) return;
    float v[8];
    float amax = 0.0f;
#pragma unroll
    for (int i = 0; i < 8; ++i) { v[i] = W[i0 + i]; amax = fmaxf(amax, fabsf(v[i])); }
    amax = fmaxf(amax, __shfl_xor(amax, 1, 4));
    amax = fmaxf(amax, __shfl_xor(amax, 2, 4));
#pragma unroll
    for (int i = 0; i < 8; ++i) Wq[i0 + i] = __float2bfloat16(mx_q(v[i], amax));
}

// ---------------- gelu (tanh approx) + MX re-quant, in place on bf16 buffer ----------------

__global__ __launch_bounds__(256)
void gelu_quant_kernel(__hip_bfloat16* __restrict__ h, size_t n) {
    size_t i0 = ((size_t)blockIdx.x * blockDim.x + threadIdx.x) * 8;
    if (i0 >= n) return;
    float g[8];
    float amax = 0.0f;
#pragma unroll
    for (int i = 0; i < 8; ++i) {
        float x = __bfloat162float(h[i0 + i]);
        float u = 0.7978845608028654f * (x + 0.044715f * x * x * x);
        g[i] = bf16r(0.5f * x * (1.0f + tanhf(u)));
        amax = fmaxf(amax, fabsf(g[i]));
    }
    amax = fmaxf(amax, __shfl_xor(amax, 1, 4));
    amax = fmaxf(amax, __shfl_xor(amax, 2, 4));
#pragma unroll
    for (int i = 0; i < 8; ++i) h[i0 + i] = __float2bfloat16(mx_q(g[i], amax));
}

// ---------------- bf16 WMMA GEMM: C[M,N] = A[M,K] * B[N,K]^T ----------------
// Macro-tile 128(M) x 256(N), 8 waves as 2(M) x 4(N), wave tile 64x64 = 4x4 WMMA accs.
// Stages double-buffered in LDS via CDNA5 async global->LDS DMA.
// MODE 0: out_bf = bf16_round(acc + bias)                               (GEMM1)
// MODE 1: out_f  = bf16_round(bf16_round(res) + bf16_round(acc + bias)) (GEMM2 + residual)

#define BM  128
#define BN  256
#define BK  32
#define LDT 40   // padded LDS row stride in halfs (80B: 16B-aligned, bank-decorrelated)

template<int MODE>
__global__ __launch_bounds__(256)
void mx_gemm_kernel(const __hip_bfloat16* __restrict__ A,
                    const __hip_bfloat16* __restrict__ B,
                    const float* __restrict__ bias,
                    const float* __restrict__ residual,
                    __hip_bfloat16* __restrict__ outBf,
                    float* __restrict__ outF,
                    int M, int N, int K) {
    __shared__ __hip_bfloat16 sA[2][BM * LDT];   // 20 KB
    __shared__ __hip_bfloat16 sB[2][BN * LDT];   // 40 KB

    const int tid  = threadIdx.x;
    const int lane = tid & 31;
    const int wave = tid >> 5;     // 0..7
    const int wm   = wave & 1;     // 2 waves along M (64 rows each)
    const int wn   = wave >> 1;    // 4 waves along N (64 cols each)
    const int m0   = blockIdx.y * BM;
    const int n0   = blockIdx.x * BN;

    v8f acc[4][4];
#pragma unroll
    for (int mi = 0; mi < 4; ++mi)
#pragma unroll
        for (int ni = 0; ni < 4; ++ni)
#pragma unroll
            for (int r = 0; r < 8; ++r) acc[mi][ni][r] = 0.0f;

    // 16B per lane per transfer; 4 lanes cover one 32-half row.
    const int rpt = tid >> 2;          // 0..63: row group
    const int col = (tid & 3) * 8;     // half offset within row

    auto load_stage = [&](int k0, int buf) {
#pragma unroll
        for (int it = 0; it < 2; ++it) {   // A: 128 rows
            const int r = rpt + it * 64;
            async_copy16((unsigned)(uintptr_t)&sA[buf][r * LDT + col],
                         A + (size_t)(m0 + r) * K + k0 + col);
        }
#pragma unroll
        for (int it = 0; it < 4; ++it) {   // B: 256 rows
            const int r = rpt + it * 64;
            async_copy16((unsigned)(uintptr_t)&sB[buf][r * LDT + col],
                         B + (size_t)(n0 + r) * K + k0 + col);
        }
    };

    auto compute_stage = [&](int buf) {
        union Frag { uint4 u[2]; v16bf v; };
        Frag afr[4], bfr[4];
        // A 16x32 bf16 layout: lanes 0-15 -> M=lane, K {0..7,16..23};
        //                      lanes16-31 -> M=lane-16, K {8..15,24..31}
        const int klo = (lane >> 4) * 8;
#pragma unroll
        for (int mi = 0; mi < 4; ++mi) {
            const int r = wm * 64 + mi * 16 + (lane & 15);
            const __hip_bfloat16* p = &sA[buf][r * LDT + klo];
            afr[mi].u[0] = *(const uint4*)(p);
            afr[mi].u[1] = *(const uint4*)(p + 16);
        }
        // B 32x16 bf16 layout: lane -> N=lane&15; lanes 0-15 K {0..7,16..23}, lanes 16-31 K {8..15,24..31}
        const int koff = (lane >> 4) * 8;
#pragma unroll
        for (int ni = 0; ni < 4; ++ni) {
            const int r = wn * 64 + ni * 16 + (lane & 15);
            const __hip_bfloat16* p = &sB[buf][r * LDT + koff];
            bfr[ni].u[0] = *(const uint4*)(p);
            bfr[ni].u[1] = *(const uint4*)(p + 16);
        }
#pragma unroll
        for (int mi = 0; mi < 4; ++mi)
#pragma unroll
            for (int ni = 0; ni < 4; ++ni)
                acc[mi][ni] = __builtin_amdgcn_wmma_f32_16x16x32_bf16(
                    false, afr[mi].v, false, bfr[ni].v,
                    (short)0, acc[mi][ni], false, false);
    };

    load_stage(0, 0);
    wait_async0();
    __syncthreads();
    const int nk = K / BK;
    for (int ks = 0; ks < nk; ++ks) {
        const int buf = ks & 1;
        if (ks + 1 < nk) load_stage((ks + 1) * BK, buf ^ 1);  // async DMA overlaps WMMAs
        compute_stage(buf);
        wait_async0();
        __syncthreads();
    }

    // Epilogue. C layout: VGPR r -> row r + (lane>>4)*8, col lane&15.
#pragma unroll
    for (int mi = 0; mi < 4; ++mi)
#pragma unroll
        for (int ni = 0; ni < 4; ++ni) {
            const int gn = n0 + wn * 64 + ni * 16 + (lane & 15);
            const int mb = m0 + wm * 64 + mi * 16 + (lane >> 4) * 8;
            const float bv = bias[gn];
#pragma unroll
            for (int r = 0; r < 8; ++r) {
                const size_t idx = (size_t)(mb + r) * N + gn;
                const float y = bf16r(acc[mi][ni][r] + bv);
                if (MODE == 0) {
                    outBf[idx] = __float2bfloat16(y);
                } else {
                    const float res = bf16r(residual[idx]);
                    outF[idx] = bf16r(res + y);
                }
            }
        }
}

// ---------------- launch ----------------

extern "C" void kernel_launch(void* const* d_in, const int* in_sizes, int n_in,
                              void* d_out, int out_size, void* d_ws, size_t ws_size,
                              hipStream_t stream) {
    const float* inp  = (const float*)d_in[0];
    const float* ln_w = (const float*)d_in[1];
    const float* ln_b = (const float*)d_in[2];
    const float* W1   = (const float*)d_in[3];
    const float* b1   = (const float*)d_in[4];
    const float* W2   = (const float*)d_in[5];
    const float* b2   = (const float*)d_in[6];
    float* out = (float*)d_out;   // reference output dtype is float32 (bf16-rounded values)

    const int H    = in_sizes[1];          // 2048
    const int H4   = in_sizes[4];          // 8192
    const int Btok = in_sizes[0] / H;      // 4096 tokens (B*S)

    char* ws = (char*)d_ws;
    __hip_bfloat16* Xq   = (__hip_bfloat16*)ws; ws += (size_t)Btok * H  * 2;
    __hip_bfloat16* W1q  = (__hip_bfloat16*)ws; ws += (size_t)H4   * H  * 2;
    __hip_bfloat16* W2q  = (__hip_bfloat16*)ws; ws += (size_t)H    * H4 * 2;
    __hip_bfloat16* Hbuf = (__hip_bfloat16*)ws; ws += (size_t)Btok * H4 * 2;

    // 1) bf16 round + LayerNorm + MX quant of activations
    ln_quant_kernel<<<Btok, 256, 0, stream>>>(inp, ln_w, ln_b, Xq, H);

    // 2) MX quant of weights (fresh every call: deterministic)
    const size_t nW = (size_t)H4 * H;
    quant_rows_kernel<<<(unsigned)((nW / 8 + 255) / 256), 256, 0, stream>>>(W1, W1q, nW);
    quant_rows_kernel<<<(unsigned)((nW / 8 + 255) / 256), 256, 0, stream>>>(W2, W2q, nW);

    // 3) GEMM1: [Btok,H] x [H4,H]^T -> bf16 [Btok,H4] (+b1, bf16 round)
    dim3 g1(H4 / BN, Btok / BM);
    mx_gemm_kernel<0><<<g1, 256, 0, stream>>>(Xq, W1q, b1, nullptr, Hbuf, nullptr,
                                              Btok, H4, H);

    // 4) gelu + MX re-quant, in place
    const size_t nH = (size_t)Btok * H4;
    gelu_quant_kernel<<<(unsigned)((nH / 8 + 255) / 256), 256, 0, stream>>>(Hbuf, nH);

    // 5) GEMM2: [Btok,H4] x [H,H4]^T -> fp32 out (+b2, bf16 round, +residual, bf16 round)
    dim3 g2(H / BN, Btok / BM);
    mx_gemm_kernel<1><<<g2, 256, 0, stream>>>(Hbuf, W2q, b2, inp, nullptr, out,
                                              Btok, H, H4);
}
